// MutualInformationLoss_73693048864814
// MI455X (gfx1250) — matrix operand order repaired and checked by reading, not compiled, not run on hardware
//
#include <hip/hip_runtime.h>

// ---------------------------------------------------------------------------
// MutualInformationLoss on gfx1250 (MI455X)
// p_xy[b] = W_pred^T (64 x Npix) @ W_tgt (Npix x 64) via v_wmma_f32_16x16x32_f16.
// Weights generated on the fly (exp), staged in LDS bin-major so fragments are
// contiguous ds_load_b128 per ISA 7.12.2 lane layouts. 128-pixel macro-chunks:
// uniform coalesced loads, one exp phase, then 16 WMMAs per barrier pair.
// ---------------------------------------------------------------------------

typedef __attribute__((ext_vector_type(16))) _Float16 v16h;
typedef __attribute__((ext_vector_type(8)))  _Float16 half8;
typedef __attribute__((ext_vector_type(2)))  _Float16 half2_t;
typedef __attribute__((ext_vector_type(8)))  float    v8f;

#define NBINS   64
#define NPIX    (512 * 512)
#define BATCH   8
#define PARTS   16               // pixel-slices per sample
#define MSTEP   128              // pixels per macro-chunk (4 WMMA K-steps)
#define CHUNK   (NPIX / PARTS)   // 16384 pixels per block
#define EPSF    1e-10f
#define INV2S2  1984.5f          // 1/(2*sigma^2), sigma = 1/63
#define INV63   (1.0f / 63.0f)

// ---------------------------------------------------------------------------
// Kernel 1: per-(sample, slice) min/max partials for pred and target
// ---------------------------------------------------------------------------
__global__ __launch_bounds__(256) void k_minmax_part(const float* __restrict__ pred,
                                                     const float* __restrict__ tgt,
                                                     float* __restrict__ mmpart) {
    const int blk  = blockIdx.x;
    const int b    = blk / PARTS;
    const int part = blk % PARTS;
    const int tid  = threadIdx.x;
    const int base = b * NPIX + part * CHUNK;

    float pmn = 3.4e38f, pmx = -3.4e38f, tmn = 3.4e38f, tmx = -3.4e38f;
    for (int i = tid * 4; i < CHUNK; i += 256 * 4) {
        float4 p = *(const float4*)&pred[base + i];
        float4 t = *(const float4*)&tgt[base + i];
        pmn = fminf(pmn, fminf(fminf(p.x, p.y), fminf(p.z, p.w)));
        pmx = fmaxf(pmx, fmaxf(fmaxf(p.x, p.y), fmaxf(p.z, p.w)));
        tmn = fminf(tmn, fminf(fminf(t.x, t.y), fminf(t.z, t.w)));
        tmx = fmaxf(tmx, fmaxf(fmaxf(t.x, t.y), fmaxf(t.z, t.w)));
    }
    __shared__ float s0[256], s1[256], s2[256], s3[256];
    s0[tid] = pmn; s1[tid] = pmx; s2[tid] = tmn; s3[tid] = tmx;
    __syncthreads();
    for (int off = 128; off > 0; off >>= 1) {
        if (tid < off) {
            s0[tid] = fminf(s0[tid], s0[tid + off]);
            s1[tid] = fmaxf(s1[tid], s1[tid + off]);
            s2[tid] = fminf(s2[tid], s2[tid + off]);
            s3[tid] = fmaxf(s3[tid], s3[tid + off]);
        }
        __syncthreads();
    }
    if (tid == 0) {
        float* o = mmpart + (b * PARTS + part) * 4;
        o[0] = s0[0]; o[1] = s1[0]; o[2] = s2[0]; o[3] = s3[0];
    }
}

// ---------------------------------------------------------------------------
// Kernel 2: fixed-order reduction of partials -> per-sample {pmin,pmax,tmin,tmax}
// ---------------------------------------------------------------------------
__global__ void k_minmax_final(const float* __restrict__ mmpart,
                               float* __restrict__ mm) {
    const int b = blockIdx.x;
    float pmn = 3.4e38f, pmx = -3.4e38f, tmn = 3.4e38f, tmx = -3.4e38f;
    for (int p = 0; p < PARTS; ++p) {
        const float* o = mmpart + (b * PARTS + p) * 4;
        pmn = fminf(pmn, o[0]); pmx = fmaxf(pmx, o[1]);
        tmn = fminf(tmn, o[2]); tmx = fmaxf(tmx, o[3]);
    }
    float* o = mm + b * 4;
    o[0] = pmn; o[1] = pmx; o[2] = tmn; o[3] = tmx;
}

// ---------------------------------------------------------------------------
// Kernel 3: WMMA joint-histogram kernel.
// Block = 128 threads = 4 waves; wave w owns row-tile w (bins 16w..16w+15) of
// the 64x64 accumulator: 4 column tiles x 4 K-steps = 16 WMMAs per macro-chunk.
// ---------------------------------------------------------------------------
__global__ __launch_bounds__(128) void k_mi_wmma(const float* __restrict__ pred,
                                                 const float* __restrict__ tgt,
                                                 const float* __restrict__ mm,
                                                 float* __restrict__ cpart) {
    const int blk  = blockIdx.x;
    const int b    = blk / PARTS;
    const int part = blk % PARTS;
    const int tid  = threadIdx.x;
    const int lane = tid & 31;
    const int wave = tid >> 5;

    // LDS: bin-major F16 weights, row = 128 halves = 256B (16B aligned rows).
    __shared__ alignas(16) _Float16 wp[NBINS * MSTEP];   // 16 KB
    __shared__ alignas(16) _Float16 wt[NBINS * MSTEP];   // 16 KB
    __shared__ float xs[2 * MSTEP];                      // 1 KB

    const float pmin = mm[b * 4 + 0], pmax = mm[b * 4 + 1];
    const float tmin = mm[b * 4 + 2], tmax = mm[b * 4 + 3];
    const float pscale = 1.0f / (pmax - pmin + EPSF);
    const float tscale = 1.0f / (tmax - tmin + EPSF);

    const int base = b * NPIX + part * CHUNK;

    // exp phase: 128 threads = 64 bins x 2 matrices
    const int   mybin = tid & 63;
    const int   mymat = tid >> 6;                  // 0 = pred, 1 = tgt
    const float c     = (float)mybin * INV63;

    v8f acc0 = {}, acc1 = {}, acc2 = {}, acc3 = {};

    const int g  = lane >> 4;     // half-wave group
    const int ml = lane & 15;     // row/col within tile

    for (int s = 0; s < CHUNK; s += MSTEP) {
        // --- phase A: uniform coalesced staging of 128 pred + 128 tgt -------
        xs[tid]         = (pred[base + s + tid] - pmin) * pscale;
        xs[MSTEP + tid] = (tgt [base + s + tid] - tmin) * tscale;
        __builtin_prefetch(&pred[base + s + MSTEP + tid], 0, 0);  // global_prefetch
        __builtin_prefetch(&tgt [base + s + MSTEP + tid], 0, 0);
        __syncthreads();

        // --- phase B: Gaussian weights, 128 exps / thread, half2 stores -----
        {
            _Float16* dst = (mymat ? wt : wp) + mybin * MSTEP;
            const float* x = &xs[mymat * MSTEP];
            #pragma unroll 16
            for (int p = 0; p < MSTEP; p += 2) {
                float d0 = x[p]     - c;
                float d1 = x[p + 1] - c;
                half2_t e = { (_Float16)__expf(-d0 * d0 * INV2S2),
                              (_Float16)__expf(-d1 * d1 * INV2S2) };
                *(half2_t*)(dst + p) = e;
            }
        }
        __syncthreads();

        // --- phase C: 4 K-steps x 4 column tiles = 16 WMMAs -----------------
        #pragma unroll
        for (int k = 0; k < 4; ++k) {
            // A (16x32 f16): lane group g holds K = k*32 + {g*8..g*8+7, +16}
            const _Float16* aptr = &wp[(wave * 16 + ml) * MSTEP + k * 32 + g * 8];
            half8 alo = *(const half8*)(aptr);
            half8 ahi = *(const half8*)(aptr + 16);
            v16h  a   = __builtin_shufflevector(alo, ahi,
                        0,1,2,3,4,5,6,7,8,9,10,11,12,13,14,15);

            // B (32x16 f16): lane = column n, lane group g holds K = k*32 + {g*8..g*8+7, +16}
            #pragma unroll
            for (int j = 0; j < 4; ++j) {
                const _Float16* bptr = &wt[(j * 16 + ml) * MSTEP + k * 32 + g * 8];
                half8 blo = *(const half8*)(bptr);
                half8 bhi = *(const half8*)(bptr + 16);
                v16h  bb  = __builtin_shufflevector(blo, bhi,
                            0,1,2,3,4,5,6,7,8,9,10,11,12,13,14,15);
                if      (j == 0) acc0 = __builtin_amdgcn_wmma_f32_16x16x32_f16(
                                     false, a, false, bb, (short)0, acc0, false, false);
                else if (j == 1) acc1 = __builtin_amdgcn_wmma_f32_16x16x32_f16(
                                     false, a, false, bb, (short)0, acc1, false, false);
                else if (j == 2) acc2 = __builtin_amdgcn_wmma_f32_16x16x32_f16(
                                     false, a, false, bb, (short)0, acc2, false, false);
                else             acc3 = __builtin_amdgcn_wmma_f32_16x16x32_f16(
                                     false, a, false, bb, (short)0, acc3, false, false);
            }
        }
        __syncthreads();  // before next macro-chunk overwrites LDS
    }

    // --- write 64x64 partial C. D layout: VGPR r, lanes 0-15: M=r, N=lane;
    //     lanes 16-31: M=r+8, N=lane-16  (+ tile offsets). ---------------------
    float* out = cpart + (b * PARTS + part) * (NBINS * NBINS);
    #pragma unroll
    for (int r = 0; r < 8; ++r) {
        const int M = wave * 16 + r + 8 * g;
        out[M * NBINS +  0 + ml] = acc0[r];
        out[M * NBINS + 16 + ml] = acc1[r];
        out[M * NBINS + 32 + ml] = acc2[r];
        out[M * NBINS + 48 + ml] = acc3[r];
    }
}

// ---------------------------------------------------------------------------
// Kernel 4: fixed-order partial reduction + marginals + MI per sample
// ---------------------------------------------------------------------------
__global__ __launch_bounds__(256) void k_mi_reduce(const float* __restrict__ cpart,
                                                   float* __restrict__ mi) {
    const int b   = blockIdx.x;
    const int tid = threadIdx.x;

    __shared__ float pxy[NBINS * NBINS];
    __shared__ float px[NBINS], py[NBINS];
    __shared__ float red[256];

    const float invN = 1.0f / (float)NPIX;
    for (int i = tid; i < NBINS * NBINS; i += 256) {
        float acc = 0.0f;
        for (int p = 0; p < PARTS; ++p)
            acc += cpart[(b * PARTS + p) * (NBINS * NBINS) + i];
        pxy[i] = acc * invN;
    }
    __syncthreads();

    if (tid < NBINS) {
        float sx = 0.0f, sy = 0.0f;
        for (int j = 0; j < NBINS; ++j) {
            sx += pxy[tid * NBINS + j];
            sy += pxy[j * NBINS + tid];
        }
        px[tid] = sx; py[tid] = sy;
    }
    __syncthreads();

    float m = 0.0f;
    for (int i = tid; i < NBINS * NBINS; i += 256) {
        float v  = pxy[i];
        float pp = px[i >> 6] * py[i & 63];
        m += v * __logf((v + EPSF) / (pp + EPSF));
    }
    red[tid] = m;
    __syncthreads();
    for (int off = 128; off > 0; off >>= 1) {
        if (tid < off) red[tid] += red[tid + off];
        __syncthreads();
    }
    if (tid == 0) mi[b] = red[0];
}

// ---------------------------------------------------------------------------
// Kernel 5: -mean over batch
// ---------------------------------------------------------------------------
__global__ void k_finalize(const float* __restrict__ mi, float* __restrict__ out) {
    float s = 0.0f;
    for (int b = 0; b < BATCH; ++b) s += mi[b];
    out[0] = -s / (float)BATCH;
}

// ---------------------------------------------------------------------------
// Workspace layout (floats):
//   [0,    512)               min/max partials (8*16*4)
//   [512,  544)               per-sample min/max (8*4)
//   [1024, 1024+524288)       C partials (8*16*64*64)
//   [525312, 525320)          per-sample MI
// Total ~2.1 MB.
// ---------------------------------------------------------------------------
extern "C" void kernel_launch(void* const* d_in, const int* in_sizes, int n_in,
                              void* d_out, int out_size, void* d_ws, size_t ws_size,
                              hipStream_t stream) {
    const float* pred = (const float*)d_in[0];
    const float* tgt  = (const float*)d_in[1];
    float* out = (float*)d_out;
    float* ws  = (float*)d_ws;

    float* mmpart = ws;                      // 512 floats
    float* mm     = ws + 512;                // 32 floats
    float* cpart  = ws + 1024;               // 524288 floats
    float* mi     = ws + 1024 + 524288;      // 8 floats

    k_minmax_part <<<BATCH * PARTS, 256, 0, stream>>>(pred, tgt, mmpart);
    k_minmax_final<<<BATCH,           1, 0, stream>>>(mmpart, mm);
    k_mi_wmma     <<<BATCH * PARTS, 128, 0, stream>>>(pred, tgt, mm, cpart);
    k_mi_reduce   <<<BATCH,         256, 0, stream>>>(cpart, mi);
    k_finalize    <<<1,               1, 0, stream>>>(mi, out);
}
